// DeformableAttention_50491635532075
// MI455X (gfx1250) — compile-verified, matrix operand order checked by reading, not run on hardware
//
#include <hip/hip_runtime.h>
#include <stdint.h>

typedef __attribute__((ext_vector_type(16))) __bf16 v16bf;
typedef __attribute__((ext_vector_type(8)))  float  v8f;
typedef __attribute__((ext_vector_type(8)))  __bf16 bf16x8;

#define Gh   8
#define KKt  9
#define Cgc  32
#define Hc   64
#define Wc   64
#define Bc   8
#define Cc   256
#define Oc   256
#define HP   66
#define WP   66
#define NP1  224          // convraw row stride in floats (216 cols used)
#define NSLICE 72         // 72 K-slices of 32
#define NSUB 16           // 16 n-subtiles of 16 (N padded to 256)

// ---------------- WMMA fragment helpers ----------------
// 16-bit A/B fragment per ISA: lane 0-15 -> row, K in {0..7,16..23};
// lane 16-31 -> same row, K in {8..15,24..31}. Dword (K-pair q) mapping:
// u[i] (i<4): q = i + 4h ; u[4+i]: q = 8 + i + 4h   (h = lane>>4)
struct Frag { union { v16bf v; uint32_t u[8]; }; };

__device__ inline Frag load_frag(const uint32_t* base, int row0) {
  int lane = threadIdx.x & 31;
  int half = lane >> 4;
  const uint32_t* r = base + (size_t)(row0 + (lane & 15)) * 16;
  Frag f;
#pragma unroll
  for (int i = 0; i < 4; ++i) {
    f.u[i]     = r[i + 4 * half];
    f.u[4 + i] = r[8 + i + 4 * half];
  }
  return f;
}

// Pre-swizzled B fragments: per (slice, nsub): 2 blocks (j=0 -> u[0..3],
// j=1 -> u[4..7]) of 32 lanes x 4 dwords (lane-major, 16B/lane, coalesced).
__device__ inline Frag load_bfrag(const uint32_t* bs, int s, int nsub) {
  int lane = threadIdx.x & 31;
  const uint4* p = (const uint4*)(bs + (size_t)(s * NSUB + nsub) * 256);
  uint4 a = p[lane];
  uint4 b = p[32 + lane];
  Frag f;
  f.u[0] = a.x; f.u[1] = a.y; f.u[2] = a.z; f.u[3] = a.w;
  f.u[4] = b.x; f.u[5] = b.y; f.u[6] = b.z; f.u[7] = b.w;
  return f;
}

union BF2 { __bf16 h[2]; uint32_t u; };

__device__ inline uint32_t pack_bf16(float lo, float hi) {
  BF2 t; t.h[0] = (__bf16)lo; t.h[1] = (__bf16)hi; return t.u;
}

// ---------------- kernel 1: pack x to padded bf16 [B][G][66][66][Cg] --------
__global__ __launch_bounds__(256) void k_pack_x(const float* __restrict__ x,
                                                __bf16* __restrict__ xp) {
  int idx = blockIdx.x * 256 + threadIdx.x;
  const int total = Bc * Gh * HP * WP * Cgc;
  if (idx >= total) return;
  int cg = idx & 31;
  int t  = idx >> 5;
  int xx = t % WP; t /= WP;
  int yy = t % HP; t /= HP;
  int g  = t & 7;  int b = t >> 3;
  float v = 0.f;
  if (yy >= 1 && yy <= Hc && xx >= 1 && xx <= Wc) {
    int c = g * Cgc + cg;
    v = x[(((size_t)b * Cc + c) * Hc + (yy - 1)) * Wc + (xx - 1)];
  }
  xp[idx] = (__bf16)v;
}

// ------- kernel 2: swizzle offset+attn weights into B fragments -------------
// conv slice s = kk*8+g covers k = kk*256 + g*32 + (0..31); col n: 0..143 off,
// 144..215 attn, rest zero.
__global__ __launch_bounds__(256) void k_pack_w1s(const float* __restrict__ woff,
                                                  const float* __restrict__ wattn,
                                                  uint32_t* __restrict__ bs) {
  int idx = blockIdx.x * 256 + threadIdx.x;
  if (idx >= NSLICE * NSUB * 2 * 32 * 4) return;
  int i    = idx & 3;
  int lane = (idx >> 2) & 31;
  int j    = (idx >> 7) & 1;
  int nsub = (idx >> 8) & 15;
  int s    = idx >> 12;
  int h = lane >> 4, n = nsub * 16 + (lane & 15);
  int q = j * 8 + i + 4 * h;           // K-pair 0..15 within slice
  int kk = s >> 3, g = s & 7;
  int c0 = g * 32 + 2 * q;
  float lo = 0.f, hi = 0.f;
  if (n < 144) {
    lo = woff[((size_t)n * Cc + c0) * KKt + kk];
    hi = woff[((size_t)n * Cc + c0 + 1) * KKt + kk];
  } else if (n < 216) {
    lo = wattn[((size_t)(n - 144) * Cc + c0) * KKt + kk];
    hi = wattn[((size_t)(n - 144) * Cc + c0 + 1) * KKt + kk];
  }
  bs[idx] = pack_bf16(lo, hi);
}

// ------- kernel 3: swizzle w_out into B fragments ---------------------------
// deform slice s = g*9+kk covers k = (g*9+kk)*32 + cg, c = g*32+cg.
__global__ __launch_bounds__(256) void k_pack_w2s(const float* __restrict__ wout,
                                                  uint32_t* __restrict__ bs) {
  int idx = blockIdx.x * 256 + threadIdx.x;
  if (idx >= NSLICE * NSUB * 2 * 32 * 4) return;
  int i    = idx & 3;
  int lane = (idx >> 2) & 31;
  int j    = (idx >> 7) & 1;
  int nsub = (idx >> 8) & 15;
  int s    = idx >> 12;
  int h = lane >> 4, n = nsub * 16 + (lane & 15);
  int q = j * 8 + i + 4 * h;
  int g = s / KKt, kk = s - g * KKt;
  int c0 = g * 32 + 2 * q;
  float lo = wout[((size_t)n * Cc + c0) * KKt + kk];
  float hi = wout[((size_t)n * Cc + c0 + 1) * KKt + kk];
  bs[idx] = pack_bf16(lo, hi);
}

// -------- kernel 4: WMMA implicit GEMM for offset+attn conv -----------------
// Tile = one image row (64 w). A tile per slice is one CONTIGUOUS 4KB region
// (channels innermost) -> async global->LDS copy, double buffered.
__global__ __launch_bounds__(256) void k_conv_gemm(const __bf16* __restrict__ xp,
                                                   const uint32_t* __restrict__ bs1,
                                                   const float* __restrict__ boff,
                                                   const float* __restrict__ battn,
                                                   float* __restrict__ convraw) {
  __shared__ uint32_t At[2][64 * 16];
  int tid = threadIdx.x, wv = tid >> 5, lane = tid & 31;
  int m0 = blockIdx.x * 64;
  int b = m0 >> 12, h = (m0 >> 6) & 63;

  auto issueA = [&](int s, int buf) {
    int kk = s >> 3, gs = s & 7;
    int ki = kk / 3, kj = kk - ki * 3;
    size_t base = (((size_t)(b * Gh + gs) * HP + (h + ki)) * WP + kj) * Cgc;
    uint64_t gaddr = (uint64_t)(uintptr_t)(xp + base + (size_t)tid * 8);
    uint32_t ldsoff = (uint32_t)(uintptr_t)(&At[buf][tid * 4]);
    asm volatile("global_load_async_to_lds_b128 %0, %1, off"
                 :: "v"(ldsoff), "v"(gaddr) : "memory");
  };

  issueA(0, 0);
  asm volatile("s_wait_asynccnt 0" ::: "memory");
  __syncthreads();

  v8f acc[8] = {};
  for (int s = 0; s < NSLICE; ++s) {
    if (s + 1 < NSLICE) issueA(s + 1, (s + 1) & 1);
    int nsub0 = wv * 2;
    if (nsub0 < 14) {            // wave 7 owns all-zero padded columns: skip
      Frag b0 = load_bfrag(bs1, s, nsub0);
      Frag b1 = load_bfrag(bs1, s, nsub0 + 1);
#pragma unroll
      for (int msub = 0; msub < 4; ++msub) {
        Frag a = load_frag(At[s & 1], msub * 16);
        acc[2 * msub]     = __builtin_amdgcn_wmma_f32_16x16x32_bf16(
            false, a.v, false, b0.v, (short)0, acc[2 * msub], false, false);
        acc[2 * msub + 1] = __builtin_amdgcn_wmma_f32_16x16x32_bf16(
            false, a.v, false, b1.v, (short)0, acc[2 * msub + 1], false, false);
      }
    }
    asm volatile("s_wait_asynccnt 0" ::: "memory");
    __syncthreads();
  }

  if (wv < 7) {
#pragma unroll
    for (int msub = 0; msub < 4; ++msub) {
#pragma unroll
      for (int nsl = 0; nsl < 2; ++nsl) {
        int n = (wv * 2 + nsl) * 16 + (lane & 15);
        if (n < 216) {
          float bias = (n < 144) ? boff[n] : battn[n - 144];
          int mrow = m0 + msub * 16 + (lane >> 4) * 8;
#pragma unroll
          for (int r = 0; r < 8; ++r)
            convraw[(size_t)(mrow + r) * NP1 + n] = acc[2 * msub + nsl][r] + bias;
        }
      }
    }
  }
}

// -------- kernel 5: softmax + bilinear sample prep --------------------------
__global__ __launch_bounds__(256) void k_prep(const float* __restrict__ convraw,
                                              float4* __restrict__ recw,
                                              uint32_t* __restrict__ recc) {
  int idx = blockIdx.x * 256 + threadIdx.x;
  if (idx >= Bc * Gh * Hc * Wc) return;
  int w = idx & 63; int t = idx >> 6;
  int h = t & 63;   t >>= 6;
  int g = t & 7;    int b = t >> 3;
  int m = (b * Hc + h) * Wc + w;
  const float* row = convraw + (size_t)m * NP1;

  float a[KKt], mx = -1e30f;
#pragma unroll
  for (int kk = 0; kk < KKt; ++kk) { a[kk] = row[144 + g * KKt + kk]; mx = fmaxf(mx, a[kk]); }
  float sum = 0.f;
#pragma unroll
  for (int kk = 0; kk < KKt; ++kk) { a[kk] = __expf(a[kk] - mx); sum += a[kk]; }
  float inv = 1.f / sum;

#pragma unroll
  for (int kk = 0; kk < KKt; ++kk) {
    float attn = a[kk] * inv;
    float dy = row[(g * KKt + kk) * 2];
    float dx = row[(g * KKt + kk) * 2 + 1];
    float py = (float)(h - 1 + kk / 3) + dy;
    float px = (float)(w - 1 + kk % 3) + dx;
    float y0f = floorf(py), x0f = floorf(px);
    float wy1 = py - y0f, wx1 = px - x0f;
    float wy0 = 1.f - wy1, wx0 = 1.f - wx1;
    // clamp padded coords into [0,65]; zero border reproduces zero-pad bilinear
    int y0 = (int)fminf(fmaxf(y0f + 1.f, 0.f), 65.f);
    int y1 = (int)fminf(fmaxf(y0f + 2.f, 0.f), 65.f);
    int x0 = (int)fminf(fmaxf(x0f + 1.f, 0.f), 65.f);
    int x1 = (int)fminf(fmaxf(x0f + 2.f, 0.f), 65.f);
    size_t p = ((size_t)((b * Gh + g) * KKt + kk) << 12) + (h * 64 + w);
    recw[p] = make_float4(attn * wy0 * wx0, attn * wy0 * wx1,
                          attn * wy1 * wx0, attn * wy1 * wx1);
    recc[p] = (uint32_t)y0 | ((uint32_t)y1 << 7) | ((uint32_t)x0 << 14) | ((uint32_t)x1 << 21);
  }
}

// -------- kernel 6: WMMA deformable GEMM with fused gather A ----------------
__global__ __launch_bounds__(256) void k_deform_gemm(const __bf16* __restrict__ xp,
                                                     const uint32_t* __restrict__ bs2,
                                                     const float4* __restrict__ recw,
                                                     const uint32_t* __restrict__ recc,
                                                     const float* __restrict__ bout,
                                                     float* __restrict__ out) {
  __shared__ uint32_t At[2][64 * 16];
  int tid = threadIdx.x, wv = tid >> 5, lane = tid & 31;
  int m0 = blockIdx.x * 64;
  int b = m0 >> 12, h = (m0 >> 6) & 63;

  auto fillA = [&](int s, int buf) {
    int g = s / KKt, kk = s - g * KKt;
    int r = tid >> 2, cg0 = (tid & 3) * 8;
    size_t p = ((size_t)((b * Gh + g) * KKt + kk) << 12) + (h * 64 + r);
    float4 wv4 = recw[p];
    uint32_t cc = recc[p];
    int y0 = cc & 127, y1 = (cc >> 7) & 127, x0 = (cc >> 14) & 127, x1 = (cc >> 21) & 127;
    const __bf16* plane = xp + (size_t)(b * Gh + g) * HP * WP * Cgc;
    bf16x8 v00 = *(const bf16x8*)(plane + ((size_t)y0 * WP + x0) * Cgc + cg0);
    bf16x8 v01 = *(const bf16x8*)(plane + ((size_t)y0 * WP + x1) * Cgc + cg0);
    bf16x8 v10 = *(const bf16x8*)(plane + ((size_t)y1 * WP + x0) * Cgc + cg0);
    bf16x8 v11 = *(const bf16x8*)(plane + ((size_t)y1 * WP + x1) * Cgc + cg0);
    bf16x8 res;
#pragma unroll
    for (int j = 0; j < 8; ++j) {
      float v = wv4.x * (float)v00[j] + wv4.y * (float)v01[j]
              + wv4.z * (float)v10[j] + wv4.w * (float)v11[j];
      res[j] = (__bf16)v;
    }
    ((bf16x8*)At[buf])[r * 4 + (tid & 3)] = res;
  };

  fillA(0, 0);
  __syncthreads();

  v8f acc[8] = {};
  for (int s = 0; s < NSLICE; ++s) {
    if (s + 1 < NSLICE) fillA(s + 1, (s + 1) & 1);
    Frag b0 = load_bfrag(bs2, s, wv * 2);
    Frag b1 = load_bfrag(bs2, s, wv * 2 + 1);
#pragma unroll
    for (int msub = 0; msub < 4; ++msub) {
      Frag a = load_frag(At[s & 1], msub * 16);
      acc[2 * msub]     = __builtin_amdgcn_wmma_f32_16x16x32_bf16(
          false, a.v, false, b0.v, (short)0, acc[2 * msub], false, false);
      acc[2 * msub + 1] = __builtin_amdgcn_wmma_f32_16x16x32_bf16(
          false, a.v, false, b1.v, (short)0, acc[2 * msub + 1], false, false);
    }
    __syncthreads();
  }

#pragma unroll
  for (int msub = 0; msub < 4; ++msub) {
#pragma unroll
    for (int nsl = 0; nsl < 2; ++nsl) {
      int j = 2 * msub + nsl;
      int o = (wv * 2 + nsl) * 16 + (lane & 15);
      float bias = bout[o];
      int mrow = m0 + msub * 16 + (lane >> 4) * 8;
      float* dst = out + ((size_t)(b * Oc + o) << 12) + (mrow & 4095);
      float4 v0 = make_float4(acc[j][0] + bias, acc[j][1] + bias,
                              acc[j][2] + bias, acc[j][3] + bias);
      float4 v1 = make_float4(acc[j][4] + bias, acc[j][5] + bias,
                              acc[j][6] + bias, acc[j][7] + bias);
      *(float4*)dst = v0;
      *(float4*)(dst + 4) = v1;
    }
  }
}

// ---------------- host launcher ----------------
extern "C" void kernel_launch(void* const* d_in, const int* in_sizes, int n_in,
                              void* d_out, int out_size, void* d_ws, size_t ws_size,
                              hipStream_t stream) {
  const float* x      = (const float*)d_in[0];
  const float* w_off  = (const float*)d_in[1];
  const float* b_off  = (const float*)d_in[2];
  const float* w_attn = (const float*)d_in[3];
  const float* b_attn = (const float*)d_in[4];
  const float* w_out  = (const float*)d_in[5];
  const float* b_out  = (const float*)d_in[6];
  float* out = (float*)d_out;

  char* ws = (char*)d_ws;
  size_t off = 0;
  auto take = [&](size_t bytes) -> char* {
    char* p = ws + off;
    off = (off + bytes + 255) & ~(size_t)255;
    return p;
  };
  __bf16*   xp   = (__bf16*)take((size_t)Bc * Gh * HP * WP * Cgc * 2);      // 17.8 MB
  uint32_t* bs1  = (uint32_t*)take((size_t)NSLICE * NSUB * 256 * 4);        // 1.2 MB
  uint32_t* bs2  = (uint32_t*)take((size_t)NSLICE * NSUB * 256 * 4);        // 1.2 MB
  float*    convraw = (float*)take((size_t)Bc * Hc * Wc * NP1 * 4);         // 29.4 MB
  float4*   recw = (float4*)take((size_t)Bc * Gh * KKt * Hc * Wc * 16);     // 37.7 MB
  uint32_t* recc = (uint32_t*)take((size_t)Bc * Gh * KKt * Hc * Wc * 4);    // 9.4 MB
  (void)ws_size; (void)in_sizes; (void)n_in; (void)out_size;

  {
    int total = Bc * Gh * HP * WP * Cgc;
    k_pack_x<<<(total + 255) / 256, 256, 0, stream>>>(x, xp);
  }
  {
    int total = NSLICE * NSUB * 2 * 32 * 4;
    k_pack_w1s<<<(total + 255) / 256, 256, 0, stream>>>(w_off, w_attn, bs1);
    k_pack_w2s<<<(total + 255) / 256, 256, 0, stream>>>(w_out, bs2);
  }
  k_conv_gemm<<<(Bc * Hc * Wc) / 64, 256, 0, stream>>>(xp, bs1, b_off, b_attn, convraw);
  {
    int total = Bc * Gh * Hc * Wc;
    k_prep<<<(total + 255) / 256, 256, 0, stream>>>(convraw, recw, recc);
  }
  k_deform_gemm<<<(Bc * Hc * Wc) / 64, 256, 0, stream>>>(xp, bs2, recw, recc, b_out, out);
}
